// AttentionHead_12695923327357
// MI455X (gfx1250) — matrix operand order repaired and checked by reading, not compiled, not run on hardware
//
#include <hip/hip_runtime.h>
#include <hip/hip_bf16.h>

// MI455X (gfx1250, wave32) fused single-head causal attention.
// Pass 0: weights fp32 -> f16.
// Pass 1: QKV projection (WMMA f16 -> f32 accum), x read exactly once (50 MB = the HBM floor).
//         Weight k-slices double-buffered into LDS via global_load_async_to_lds_b128 (ASYNCcnt),
//         deduplicating the 4 waves' B-fragment reads and overlapping load with WMMA.
//         Q,K stored row-major f16; V stored transposed Vt[b][h][s] f16 (L2-resident, 6.6 MB total).
// Pass 2: flash attention, one wave per 16-query tile, 32-key steps, online softmax,
//         scores & PV via v_wmma_f32_16x16x32_f16; P relayout C->A via per-wave LDS.

#define NB 8
#define SEQ 2048
#define EMB 768
#define HD 64
#define WROWS 192        // 3 matrices * 64 output cols
#define WSTRIDE 40       // 32 k-halves + 8 pad (bank-conflict avoidance)

typedef _Float16 v16h __attribute__((ext_vector_type(16)));
typedef _Float16 v8h  __attribute__((ext_vector_type(8)));
typedef float    v8f  __attribute__((ext_vector_type(8)));

union F16x16 { v16h v; v8h h[2]; };

// Build a 16-half fragment from two 16B groups.
__device__ inline v16h ldA(const _Float16* p0, const _Float16* p1) {
  F16x16 u;
  u.h[0] = *(const v8h*)p0;
  u.h[1] = *(const v8h*)p1;
  return u.v;
}

// ---------------- Pass 0: weight conversion fp32 -> f16, packed [3][64][768] ----------------
__global__ void cvt_w_kernel(const float* __restrict__ Wq, const float* __restrict__ Wk,
                             const float* __restrict__ Wv, _Float16* __restrict__ Wh) {
  int t = blockIdx.x * 256 + threadIdx.x;
  if (t >= 3 * HD * EMB) return;
  int m = t / (HD * EMB), i = t % (HD * EMB);
  const float* w = (m == 0) ? Wq : ((m == 1) ? Wk : Wv);
  Wh[t] = (_Float16)w[i];
}

// Issue async copy of one 32-wide K-slice of all weights into LDS buffer `buf`.
// 192 rows x 64B = 768 x 16B chunks, 128 threads x 6 chunks each. Tracked by ASYNCcnt.
__device__ inline void issue_wslice(const _Float16* __restrict__ Wh, _Float16* wlds,
                                    int buf, int k0, int tid) {
  _Float16* base = wlds + buf * (WROWS * WSTRIDE);
#pragma unroll
  for (int i = 0; i < 6; ++i) {
    int ch = i * 128 + tid;            // chunk id in [0, 768)
    int row = ch >> 2;                 // [0, 192)
    int part = ch & 3;                 // 4 x 16B per row
    unsigned lds_off = (unsigned)(uintptr_t)(base + row * WSTRIDE + part * 8);
    const _Float16* src = Wh + row * EMB + k0 + part * 8;
    asm volatile("global_load_async_to_lds_b128 %0, %1, off"
                 :: "v"(lds_off), "v"(src) : "memory");
  }
}

// ---------------- Pass 1: QKV projection ----------------
// 4 waves per block, each wave owns a 16-row M-tile. N=64 (4 n-tiles), K=768 (24 k-steps).
__global__ __launch_bounds__(128) void qkv_proj_kernel(
    const float* __restrict__ x, const _Float16* __restrict__ Wh,
    _Float16* __restrict__ Qh, _Float16* __restrict__ Kh, _Float16* __restrict__ Vt) {
  __shared__ _Float16 wlds[2 * WROWS * WSTRIDE];  // double-buffered weight slice (30 KB)
  __shared__ _Float16 stage[4 * 16 * HD];         // per-wave 16x64 staging tile (8 KB)
  const int tid = threadIdx.x;
  const int l   = tid & 31;
  const int wid = tid >> 5;
  const int lr  = l & 15;        // row (A) / col (B,C)
  const int hi  = l >> 4;        // lane-half
  const int kb  = hi * 8;        // unused (A-frag uses koff too)
  const int koff = hi * 16;      // B-frag contiguous K base
  const int tile = blockIdx.x * 4 + wid;          // 1024 M-tiles total
  const long row0 = (long)tile * 16;
  const int b  = tile >> 7;                        // 128 tiles per batch
  const int s0 = (tile & 127) * 16;

  const float* xrow = x + (row0 + lr) * EMB;

  v8f zero = {};
  v8f acc[12];
#pragma unroll
  for (int i = 0; i < 12; ++i) acc[i] = zero;

  // prime buffer 0 with k-slice 0
  issue_wslice(Wh, wlds, 0, 0, tid);
  asm volatile("s_wait_asynccnt 0" ::: "memory");
  __syncthreads();

  int cur = 0;
  for (int k0 = 0; k0 < EMB; k0 += 32) {
    if (k0 + 32 < EMB) issue_wslice(Wh, wlds, cur ^ 1, k0 + 32, tid);
    __builtin_prefetch(xrow + k0 + 160, 0, 0);  // stream-ahead on the HBM-bound x read

    // A fragment: x rows, fp32 -> f16
    v16h a;
    const float* p0 = xrow + k0 + koff;
    const float* p1 = p0 + 8;
#pragma unroll
    for (int i = 0; i < 8; ++i) { a[i] = (_Float16)p0[i]; a[8 + i] = (_Float16)p1[i]; }

    const _Float16* wcur = wlds + cur * (WROWS * WSTRIDE);
#pragma unroll
    for (int m = 0; m < 3; ++m) {
#pragma unroll
      for (int nt = 0; nt < 4; ++nt) {
        const _Float16* wp = wcur + (m * HD + nt * 16 + lr) * WSTRIDE + koff;
        v16h bf = ldA(wp, wp + 8);
        acc[m * 4 + nt] = __builtin_amdgcn_wmma_f32_16x16x32_f16(
            false, a, false, bf, (short)0, acc[m * 4 + nt], false, false);
      }
    }
    asm volatile("s_wait_asynccnt 0" ::: "memory");
    __syncthreads();
    cur ^= 1;
  }

  // Q, K: C-frag -> LDS tile -> contiguous 16B global stores (row-major [B*S, 64] f16)
  _Float16* st = stage + wid * (16 * HD);
#pragma unroll
  for (int m = 0; m < 2; ++m) {
#pragma unroll
    for (int nt = 0; nt < 4; ++nt)
#pragma unroll
      for (int j = 0; j < 8; ++j)
        st[(j + hi * 8) * HD + nt * 16 + lr] = (_Float16)acc[m * 4 + nt][j];
    // DS ops are in-order within a wave: loads below see the stores above.
    _Float16* dst = ((m == 0) ? Qh : Kh) + row0 * HD;
#pragma unroll
    for (int i = 0; i < 4; ++i) {
      int e = l * 32 + i * 8;  // 32 lanes * 4 * 8 halves = 1024
      *(v8h*)(dst + e) = *(const v8h*)(st + e);
    }
  }

  // V: store transposed Vt[b][h][s]; C-frag rows are contiguous along s -> packed 16B stores
#pragma unroll
  for (int nt = 0; nt < 4; ++nt) {
    v8h pk;
#pragma unroll
    for (int j = 0; j < 8; ++j) pk[j] = (_Float16)acc[8 + nt][j];
    int h = nt * 16 + lr;
    *(v8h*)(Vt + ((long)b * HD + h) * SEQ + s0 + hi * 8) = pk;
  }
}

// ---------------- Pass 2: flash attention ----------------
// 4 waves per block; each wave owns a 16-query tile, iterates keys in steps of 32.
__global__ __launch_bounds__(128) void flash_attn_kernel(
    const _Float16* __restrict__ Qh, const _Float16* __restrict__ Kh,
    const _Float16* __restrict__ Vt, float* __restrict__ out) {
  __shared__ _Float16 plds[4 * 16 * 40];  // per-wave 16x32 P tile, row stride 40 halves (pad)
  const int l   = threadIdx.x & 31;
  const int wid = threadIdx.x >> 5;
  const int lr  = l & 15;
  const int hi  = l >> 4;
  const int kb  = hi * 8;
  const int koff = hi * 16;
  const int qt = blockIdx.x * 4 + wid;
  const int b  = qt >> 7;
  const int qb = (qt & 127) * 16;
  const long base = (long)b * SEQ;
  const float SCALE = 27.712812921102035f;  // sqrt(768), faithful to reference

  // Q A-fragments (dims 0-31 and 32-63), reused across the whole key loop
  const _Float16* Qrow = Qh + (base + qb + lr) * HD;
  const v16h aq0 = ldA(Qrow + koff,      Qrow + 8 + koff);
  const v16h aq1 = ldA(Qrow + 32 + koff, Qrow + 40 + koff);

  float mrow[8], lrow[8], corr[8];
  v8f zero = {};
  v8f o[4];
#pragma unroll
  for (int nt = 0; nt < 4; ++nt) o[nt] = zero;
#pragma unroll
  for (int j = 0; j < 8; ++j) { mrow[j] = -1e30f; lrow[j] = 0.f; }

  _Float16* P = plds + wid * (16 * 40);
  const int nsteps = (qb + 47) >> 5;  // keys [0, nsteps*32) cover all k <= qb+15

  for (int s = 0; s < nsteps; ++s) {
    const int k32 = s * 32;
    if (s + 1 < nsteps) {  // stream-ahead hints for next step's K/V tiles
      __builtin_prefetch(Kh + (base + k32 + 32 + lr) * HD + koff, 0, 0);
      __builtin_prefetch(Vt + ((long)b * HD + lr) * SEQ + k32 + 32, 0, 0);
    }
    // ---- scores: two 16x16 tiles (keys k32..+15, k32+16..+31), K-dim 64 in two wmma each
    v8f s0 = zero, s1 = zero;
    {
      const _Float16* kp0 = Kh + (base + k32 + lr) * HD + koff;       // key rows = B columns
      const _Float16* kp1 = kp0 + 16 * HD;
      s0 = __builtin_amdgcn_wmma_f32_16x16x32_f16(false, aq0, false, *(const v16h*)kp0,        (short)0, s0, false, false);
      s0 = __builtin_amdgcn_wmma_f32_16x16x32_f16(false, aq1, false, *(const v16h*)(kp0 + 32), (short)0, s0, false, false);
      s1 = __builtin_amdgcn_wmma_f32_16x16x32_f16(false, aq0, false, *(const v16h*)kp1,        (short)0, s1, false, false);
      s1 = __builtin_amdgcn_wmma_f32_16x16x32_f16(false, aq1, false, *(const v16h*)(kp1 + 32), (short)0, s1, false, false);
    }
#pragma unroll
    for (int j = 0; j < 8; ++j) { s0[j] *= SCALE; s1[j] *= SCALE; }
    if (k32 + 31 > qb) {  // causal mask (wave-uniform branch, last step(s) only)
#pragma unroll
      for (int j = 0; j < 8; ++j) {
        int q = qb + j + hi * 8;
        int kk = k32 + lr;
        if (kk > q)      s0[j] = -1e30f;
        if (kk + 16 > q) s1[j] = -1e30f;
      }
    }
    // ---- online softmax: per-row max/sum via xor-shuffles within 16-lane halves
    v8f p0, p1;
#pragma unroll
    for (int j = 0; j < 8; ++j) {
      float rv = fmaxf(s0[j], s1[j]);
#pragma unroll
      for (int off = 8; off >= 1; off >>= 1) rv = fmaxf(rv, __shfl_xor(rv, off, 32));
      float mnew = fmaxf(mrow[j], rv);
      float c  = __expf(mrow[j] - mnew);
      float e0 = __expf(s0[j] - mnew);
      float e1 = __expf(s1[j] - mnew);
      float rs = e0 + e1;
#pragma unroll
      for (int off = 8; off >= 1; off >>= 1) rs += __shfl_xor(rs, off, 32);
      lrow[j] = lrow[j] * c + rs;
      mrow[j] = mnew;
      corr[j] = c;
      p0[j] = e0; p1[j] = e1;
    }
#pragma unroll
    for (int nt = 0; nt < 4; ++nt)
#pragma unroll
      for (int j = 0; j < 8; ++j) o[nt][j] *= corr[j];
    // ---- P: C-layout -> LDS -> A-layout (DS in-order within wave)
#pragma unroll
    for (int j = 0; j < 8; ++j) {
      int r = j + hi * 8;
      P[r * 40 + lr]      = (_Float16)p0[j];
      P[r * 40 + 16 + lr] = (_Float16)p1[j];
    }
    const _Float16* pr = P + lr * 40;
    v16h ap = ldA(pr + koff, pr + 8 + koff);
    // ---- P(16x32) x V(32x64): V columns contiguous in Vt
#pragma unroll
    for (int nt = 0; nt < 4; ++nt) {
      const _Float16* vp = Vt + ((long)b * HD + nt * 16 + lr) * SEQ + k32 + koff;
      o[nt] = __builtin_amdgcn_wmma_f32_16x16x32_f16(
          false, ap, false, *(const v16h*)vp, (short)0, o[nt], false, false);
    }
  }

  // epilogue: normalize and store fp32 [B, S, 64]
#pragma unroll
  for (int nt = 0; nt < 4; ++nt)
#pragma unroll
    for (int j = 0; j < 8; ++j)
      out[(base + qb + j + hi * 8) * HD + nt * 16 + lr] = o[nt][j] / lrow[j];
}

extern "C" void kernel_launch(void* const* d_in, const int* in_sizes, int n_in,
                              void* d_out, int out_size, void* d_ws, size_t ws_size,
                              hipStream_t stream) {
  const float* x  = (const float*)d_in[0];
  const float* Wq = (const float*)d_in[1];
  const float* Wk = (const float*)d_in[2];
  const float* Wv = (const float*)d_in[3];
  float* out = (float*)d_out;

  // workspace layout (f16 elements): Wh[3*64*768] | Qh[16384*64] | Kh | Vt  (~6.6 MB)
  _Float16* wsh = (_Float16*)d_ws;
  _Float16* Wh = wsh;
  _Float16* Qh = Wh + 3 * HD * EMB;
  _Float16* Kh = Qh + (long)NB * SEQ * HD;
  _Float16* Vt = Kh + (long)NB * SEQ * HD;

  cvt_w_kernel<<<(3 * HD * EMB + 255) / 256, 256, 0, stream>>>(Wq, Wk, Wv, Wh);
  qkv_proj_kernel<<<(NB * SEQ / 16) / 4, 128, 0, stream>>>(x, Wh, Qh, Kh, Vt);
  flash_attn_kernel<<<(NB * SEQ / 16) / 4, 128, 0, stream>>>(Qh, Kh, Vt, out);
}
